// S_Attention_11802570130231
// MI455X (gfx1250) — compile-verified, matrix operand order checked by reading, not run on hardware
//
#include <hip/hip_runtime.h>
#include <hip/hip_bf16.h>
#include <math.h>

// ---------------------------------------------------------------------------
// Problem constants (from reference): S=256, W=128, H=768, TOPK=3, L=3*W=384.
// Output: (256, 127, 768) f32.
// ---------------------------------------------------------------------------
#define S_DIM 256
#define W_DIM 128
#define H_DIM 768
#define L_DIM 384   // TOPK * W
#define OUTW  127   // W - 1

typedef __attribute__((ext_vector_type(16))) _Float16 v16h;
typedef __attribute__((ext_vector_type(8)))  _Float16 v8h;
typedef __attribute__((ext_vector_type(8)))  float    v8f;

// ---------------------------------------------------------------------------
// CDNA5 async global->LDS copy (ASYNCcnt-tracked).  LDS address = low 32 bits
// of the generic pointer (ISA: LDS aperture maps addr[31:0] -> LDS_ADDR).
// ---------------------------------------------------------------------------
__device__ __forceinline__ unsigned lds_addr_of(const void* p) {
    return (unsigned)(unsigned long long)p;
}

__device__ __forceinline__ void async_b128(unsigned lds_off, const void* gaddr) {
    asm volatile("global_load_async_to_lds_b128 %0, %1, off"
                 :: "v"(lds_off), "v"(gaddr) : "memory");
}

// Copy `BYTES` from global to LDS with NT threads (16B per lane per op).
template <int BYTES, int NT>
__device__ __forceinline__ void async_copy_tile(_Float16* lds_dst,
                                                const _Float16* gsrc, int tid) {
    const unsigned l0 = lds_addr_of(lds_dst);
    const char*    g  = (const char*)gsrc;
    #pragma unroll
    for (int off = 0; off < BYTES; off += NT * 16) {
        const int o = off + tid * 16;
        async_b128(l0 + (unsigned)o, g + o);
    }
}

#define WAIT_ASYNC(n) asm volatile("s_wait_asynccnt %0" :: "n"(n) : "memory")

// ---------------------------------------------------------------------------
// Kernel 1: per-row L1 distances on x0 = inputs[:,0,:] and top-3 smallest
// (stable tie-break on smallest index, matching argsort). One block per s.
// ---------------------------------------------------------------------------
__global__ void __launch_bounds__(256) topk_kernel(const float* __restrict__ inputs,
                                                   int* __restrict__ idx_out) {
    __shared__ float xs[H_DIM];
    __shared__ float dists[S_DIM];

    const int s   = blockIdx.x;
    const int tid = threadIdx.x;

    for (int h = tid; h < H_DIM; h += 256)
        xs[h] = inputs[(size_t)s * W_DIM * H_DIM + h];
    __syncthreads();

    const float* xj = inputs + (size_t)tid * W_DIM * H_DIM;
    float d = 0.0f;
    #pragma unroll 8
    for (int h = 0; h < H_DIM; ++h)
        d += fabsf(xs[h] - xj[h]);
    dists[tid] = d;
    __syncthreads();

    if (tid == 0) {
        int chosen[3] = {-1, -1, -1};
        for (int k = 0; k < 3; ++k) {
            float bd = 3.4e38f;
            int   bj = 0;
            for (int j = 0; j < S_DIM; ++j) {
                if (j == chosen[0] || j == chosen[1] || j == chosen[2]) continue;
                float dj = dists[j];
                if (dj < bd) { bd = dj; bj = j; }   // strict < == stable argsort
            }
            chosen[k] = bj;
            idx_out[s * 3 + k] = bj;
        }
    }
}

// ---------------------------------------------------------------------------
// Kernel 2: build x_f16[s][l][h] = f16( inputs[idx[s][l/128]][l%128][h]+PE(l,h) )
// and its transpose xT_f16[s][h][l], both written coalesced via an LDS tile.
// ---------------------------------------------------------------------------
__global__ void __launch_bounds__(256) build_x_kernel(const float* __restrict__ inputs,
                                                      const int* __restrict__ idxb,
                                                      _Float16* __restrict__ xf16,
                                                      _Float16* __restrict__ xtf16) {
    __shared__ _Float16 tile[64][65];   // +1 pad against bank conflicts

    const int s   = blockIdx.z;
    const int l0  = blockIdx.y * 64;
    const int h0  = blockIdx.x * 64;
    const int tid = threadIdx.x;

    const float LN1E4_OVER_H = 9.210340371976184f / 768.0f;

    #pragma unroll
    for (int e = 0; e < 16; ++e) {
        const int flat = e * 256 + tid;          // 0..4095
        const int hl = flat & 63, ll = flat >> 6;
        const int l = l0 + ll, h = h0 + hl;
        const int sp = idxb[s * 3 + (l >> 7)];
        const float v = inputs[((size_t)sp * W_DIM + (l & (W_DIM - 1))) * H_DIM + h];
        const float freq  = __expf(-(float)(h & ~1) * LN1E4_OVER_H);
        const float angle = (float)l * freq;
        const float pe    = (h & 1) ? __cosf(angle) : __sinf(angle);
        const _Float16 r  = (_Float16)(v + pe);
        xf16[((size_t)s * L_DIM + l) * H_DIM + h] = r;   // coalesced along h
        tile[ll][hl] = r;
    }
    __syncthreads();

    #pragma unroll
    for (int e = 0; e < 16; ++e) {
        const int flat = e * 256 + tid;
        const int ll = flat & 63, hl = flat >> 6;
        xtf16[((size_t)s * H_DIM + h0 + hl) * L_DIM + l0 + ll] = tile[ll][hl]; // coalesced along l
    }
}

// ---------------------------------------------------------------------------
// WMMA fragment loader (16-bit operands, wave32), per ISA 7.12.2 layout.
//   lane l: m/n = l & 15, group g = l >> 4
//   halves 0..7  = k0 + g*8 + 0..7      (contiguous 16 B)
//   halves 8..15 = k0 + g*8 + 16..23    (contiguous 16 B)
// Two 16B vector loads -> ds_load_b128 / global_load_b128.
// ---------------------------------------------------------------------------
__device__ __forceinline__ v16h load_frag_rowk(const _Float16* __restrict__ base,
                                               int row0, int k0, int ld) {
    const int lane = threadIdx.x & 31;
    const int g    = lane >> 4;
    const _Float16* p = base + (size_t)(row0 + (lane & 15)) * ld + k0 + g * 8;
    union { v16h v; v8h q[2]; } u;
    u.q[0] = *(const v8h*)(p);
    u.q[1] = *(const v8h*)(p + 16);
    return u.v;
}

// ---------------------------------------------------------------------------
// Kernel 3: attention, one block = one s, 256 threads = 8 waves.
// Wave w owns query rows w*16 .. w*16+15 (covers all 128 needed rows).
// Phase 1: scores = (x_q . x^T) * 1/sqrt(H); 16-row kv tiles async-copied into
//          double-buffered LDS, consumed by all 8 waves; B frags pipelined.
// Phase 2: row softmax in LDS; f16 probabilities written IN PLACE into the
//          f32 score rows (half j at bytes 2j < float j at 4j -> WAR-safe).
// Phase 3: out = P . x via xT tiles, same async double-buffering.
// Dynamic LDS: 128*384*4 (scores/probs) + 2*24576 (tiles) = 245760 B.
// ---------------------------------------------------------------------------
#define TILE1_BYTES (16 * H_DIM * 2)   // 24576: kv tile -> 6 async per wave
#define TILE3_BYTES (16 * L_DIM * 2)   // 12288: h tile  -> 3 async per wave
#define TILE_STRIDE_H 12288            // halves between the two LDS buffers
#define SC_STRIDE 384                  // f32 score row stride (floats)
#define P_STRIDE  768                  // same rows viewed as f16 (halves)

__global__ void __launch_bounds__(256) attn_kernel(const _Float16* __restrict__ xall,
                                                   const _Float16* __restrict__ xtall,
                                                   float* __restrict__ out) {
    extern __shared__ char smem[];
    float*    s_sc = (float*)smem;                            // [128][384] f32
    _Float16* s_bt = (_Float16*)(smem + 128 * SC_STRIDE * 4); // 2 x tile buffers

    const int s    = blockIdx.x;
    const int tid  = threadIdx.x;
    const int wave = tid >> 5;
    const int lane = tid & 31;

    const _Float16* x  = xall  + (size_t)s * L_DIM * H_DIM;
    const _Float16* xt = xtall + (size_t)s * H_DIM * L_DIM;

    // ---- Phase 1: scores (wave-private 16 query rows x all 384 kv cols) ----
    const int m0 = wave * 16;

    v16h afr[24];                                    // query A-frags, resident
    #pragma unroll
    for (int kb = 0; kb < 24; ++kb)
        afr[kb] = load_frag_rowk(x, m0, kb * 32, H_DIM);

    const float scale = 0.03608439182435161f;        // 1/sqrt(768)

    async_copy_tile<TILE1_BYTES, 256>(s_bt, x, tid); // prefetch kv tile 0
    for (int c = 0; c < 24; ++c) {
        if (c + 1 < 24)
            async_copy_tile<TILE1_BYTES, 256>(s_bt + ((c + 1) & 1) * TILE_STRIDE_H,
                                              x + (size_t)(c + 1) * 16 * H_DIM, tid);
        if (c + 1 < 24) { WAIT_ASYNC(6); } else { WAIT_ASYNC(0); }
        __syncthreads();                             // tile c visible to all waves

        const _Float16* bt = s_bt + (c & 1) * TILE_STRIDE_H;
        v8f acc = {};
        v16h bcur = load_frag_rowk(bt, 0, 0, H_DIM);
        #pragma unroll
        for (int kb = 0; kb < 24; ++kb) {            // pipelined B frags
            v16h bnext;
            if (kb + 1 < 24) bnext = load_frag_rowk(bt, 0, (kb + 1) * 32, H_DIM);
            acc = __builtin_amdgcn_wmma_f32_16x16x32_f16(
                false, afr[kb], false, bcur, (short)0, acc, false, false);
            bcur = bnext;
        }
        const int n     = c * 16 + (lane & 15);
        const int rbase = wave * 16 + 8 * (lane >> 4);
        #pragma unroll
        for (int r = 0; r < 8; ++r)
            s_sc[(rbase + r) * SC_STRIDE + n] = acc[r] * scale;
        __syncthreads();                             // tile c consumed; buffer reusable
    }

    // ---- Phase 2: softmax over 384 cols; one thread per row (128 rows) ----
    if (tid < 128) {
        float* row = s_sc + tid * SC_STRIDE;
        float mx = row[0];
        for (int j = 1; j < L_DIM; ++j) mx = fmaxf(mx, row[j]);
        float sum = 0.0f;
        for (int j = 0; j < L_DIM; ++j) { float e = __expf(row[j] - mx); row[j] = e; sum += e; }
        const float inv = 1.0f / sum;
        _Float16* prow = (_Float16*)row;             // in-place f32 -> f16 (WAR-safe)
        for (int j = 0; j < L_DIM; ++j) { float e = row[j]; prow[j] = (_Float16)(e * inv); }
    }
    __syncthreads();

    // ---- Phase 3: out = P (16x384 f16 in LDS) @ x (384x768) via xT tiles ----
    const _Float16* s_p = (const _Float16*)s_sc;     // rows at stride P_STRIDE halves
    v16h pfr[12];
    #pragma unroll
    for (int kb = 0; kb < 12; ++kb)
        pfr[kb] = load_frag_rowk(s_p, wave * 16, kb * 32, P_STRIDE);

    async_copy_tile<TILE3_BYTES, 256>(s_bt, xt, tid);   // prefetch h tile 0
    for (int hg = 0; hg < 48; ++hg) {
        if (hg + 1 < 48)
            async_copy_tile<TILE3_BYTES, 256>(s_bt + ((hg + 1) & 1) * TILE_STRIDE_H,
                                              xt + (size_t)(hg + 1) * 16 * L_DIM, tid);
        if (hg + 1 < 48) { WAIT_ASYNC(3); } else { WAIT_ASYNC(0); }
        __syncthreads();

        const _Float16* bt = s_bt + (hg & 1) * TILE_STRIDE_H;
        v8f acc = {};
        v16h bcur = load_frag_rowk(bt, 0, 0, L_DIM);
        #pragma unroll
        for (int kb = 0; kb < 12; ++kb) {            // pipelined B frags
            v16h bnext;
            if (kb + 1 < 12) bnext = load_frag_rowk(bt, 0, (kb + 1) * 32, L_DIM);
            acc = __builtin_amdgcn_wmma_f32_16x16x32_f16(
                false, pfr[kb], false, bcur, (short)0, acc, false, false);
            bcur = bnext;
        }
        const int n     = hg * 16 + (lane & 15);
        const int rbase = wave * 16 + 8 * (lane >> 4);
        #pragma unroll
        for (int r = 0; r < 8; ++r) {
            const int g = rbase + r;                 // global query row
            if (g < OUTW)
                out[((size_t)s * OUTW + g) * H_DIM + n] = acc[r];
        }
        __syncthreads();
    }
}

// ---------------------------------------------------------------------------
// Host launch
// ---------------------------------------------------------------------------
extern "C" void kernel_launch(void* const* d_in, const int* in_sizes, int n_in,
                              void* d_out, int out_size, void* d_ws, size_t ws_size,
                              hipStream_t stream) {
    const float* inputs = (const float*)d_in[0];
    float*       out    = (float*)d_out;

    // ws layout: [0,3072) idx i32[256][3];
    //            [4096, +151MB) x_f16[s][l][h];
    //            then xT_f16[s][h][l] (+151MB)
    const size_t X_BYTES = (size_t)S_DIM * L_DIM * H_DIM * sizeof(_Float16);
    int*      idxb  = (int*)d_ws;
    _Float16* xf16  = (_Float16*)((char*)d_ws + 4096);
    _Float16* xtf16 = (_Float16*)((char*)d_ws + 4096 + X_BYTES);

    topk_kernel<<<S_DIM, 256, 0, stream>>>(inputs, idxb);
    build_x_kernel<<<dim3(H_DIM / 64, L_DIM / 64, S_DIM), 256, 0, stream>>>(
        inputs, idxb, xf16, xtf16);

    const size_t smem = (size_t)128 * SC_STRIDE * 4 + 2 * TILE1_BYTES;  // 245760 B
    attn_kernel<<<S_DIM, 256, smem, stream>>>(xf16, xtf16, out);
}
